// MultiHeadAttention_68667937128978
// MI455X (gfx1250) — matrix operand order repaired and checked by reading, not compiled, not run on hardware
//
#include <hip/hip_runtime.h>
#include <hip/hip_bf16.h>

#define SEQ  2048
#define EDIM 1024
#define HD   64
#define NH   16
#define NB   2

typedef __attribute__((ext_vector_type(16))) __bf16 v16bf;
typedef __attribute__((ext_vector_type(8)))  float  v8f;

struct U8 { unsigned int u[8]; };

__device__ __forceinline__ v16bf as_v16bf(const U8& x) {
  union { U8 a; v16bf b; } c; c.a = x; return c.b;
}

// round-half-up fp32 -> bf16
__device__ __forceinline__ unsigned short f2bf(float f) {
  unsigned int x = __builtin_bit_cast(unsigned int, f);
  return (unsigned short)((x + 0x8000u) >> 16);
}
// pack two fp32 -> bf16x2 with one v_perm_b32
__device__ __forceinline__ unsigned int pack2(float lo, float hi) {
  unsigned int rl = __builtin_bit_cast(unsigned int, lo) + 0x8000u;
  unsigned int rh = __builtin_bit_cast(unsigned int, hi) + 0x8000u;
  return __builtin_amdgcn_perm(rh, rl, 0x07060302u);
}

// A-operand (16x32 bf16): lane holds row (lane&15); K runs are
// [half*8 .. +7] and [16+half*8 .. +7]  -> two 16B chunks.
__device__ __forceinline__ void loadA(const unsigned short* rowp, int half, U8& a) {
  uint4 lo = *(const uint4*)(rowp + half * 8);
  uint4 hi = *(const uint4*)(rowp + 16 + half * 8);
  a.u[0] = lo.x; a.u[1] = lo.y; a.u[2] = lo.z; a.u[3] = lo.w;
  a.u[4] = hi.x; a.u[5] = hi.y; a.u[6] = hi.z; a.u[7] = hi.w;
}
// B-operand (32x16 bf16): lane holds col (lane&15); K runs are
// [half*8 .. +7] and [16+half*8 .. +7]  -> two 16B chunks.
__device__ __forceinline__ void loadB(const unsigned short* rowp, int half, U8& b) {
  uint4 lo = *(const uint4*)(rowp + half * 8);
  uint4 hi = *(const uint4*)(rowp + 16 + half * 8);
  b.u[0] = lo.x; b.u[1] = lo.y; b.u[2] = lo.z; b.u[3] = lo.w;
  b.u[4] = hi.x; b.u[5] = hi.y; b.u[6] = hi.z; b.u[7] = hi.w;
}

__device__ __forceinline__ v8f wmma_bf16(const U8& a, const U8& b, v8f c) {
  return __builtin_amdgcn_wmma_f32_16x16x32_bf16(
      false, as_v16bf(a), false, as_v16bf(b), (short)0, c, false, false);
}

// async copy: each lane moves 16B global -> LDS (tracked by ASYNCcnt)
__device__ __forceinline__ void async_g2l_b128(unsigned lds_addr, const void* gaddr) {
  asm volatile("global_load_async_to_lds_b128 %0, %1, off"
               :: "v"(lds_addr), "v"((unsigned long long)(size_t)gaddr)
               : "memory");
}
__device__ __forceinline__ unsigned lds_off(const void* p) {
  return (unsigned)(size_t)p;   // generic->LDS: low 32 bits = group-segment offset
}

// ---------------------------------------------------------------------------
// Kernel 0: one-shot fp32 -> bf16 conversion (8 elements / thread)
// ---------------------------------------------------------------------------
__global__ void cvt_f32_to_bf16(const float* __restrict__ src,
                                unsigned short* __restrict__ dst, int n) {
  int i = (blockIdx.x * blockDim.x + threadIdx.x) * 8;
  if (i >= n) return;
  float4 a = *(const float4*)(src + i);
  float4 b = *(const float4*)(src + i + 4);
  uint4 o;
  o.x = pack2(a.x, a.y);
  o.y = pack2(a.z, a.w);
  o.z = pack2(b.x, b.y);
  o.w = pack2(b.z, b.w);
  *(uint4*)(dst + i) = o;
}

// ---------------------------------------------------------------------------
// Kernel 1: fused Q/K/V projection, bf16, W staged in LDS via async copy.
// grid=(16,16,3), block=256 (8 waves). wave -> 32 rows x 64 cols (one head).
// __launch_bounds__(256,1): budget VGPRs for 2 waves/SIMD -> no acc spills.
// ---------------------------------------------------------------------------
__global__ void __launch_bounds__(256, 1)
mha_qkv_proj(const unsigned short* __restrict__ xq,
             const unsigned short* __restrict__ xk,
             const unsigned short* __restrict__ xv,
             const unsigned short* __restrict__ wq,
             const unsigned short* __restrict__ wk,
             const unsigned short* __restrict__ wv,
             unsigned short* __restrict__ qh,
             unsigned short* __restrict__ kh,
             unsigned short* __restrict__ vt) {
  __shared__ __align__(16) unsigned short bsm[2][64 * 32];   // 8KB

  const int tid  = threadIdx.x;
  const int lane = tid & 31, wave = tid >> 5;
  const int rrow = lane & 15, half = lane >> 4;
  const int row0 = (blockIdx.x * 8 + wave) * 32;
  const int h    = blockIdx.y;
  const int col0 = h * HD;
  const int which = blockIdx.z;

  const unsigned short* x = (which == 0) ? xq : (which == 1) ? xk : xv;
  const unsigned short* W = (which == 0) ? wq : (which == 1) ? wk : wv;

  const int crow = tid >> 2, cofs = (tid & 3) * 8;
  const unsigned short* wsrc = W + (size_t)(col0 + crow) * EDIM + cofs;
  const unsigned ldst0 = lds_off(&bsm[0][crow * 32 + cofs]);
  const unsigned ldst1 = lds_off(&bsm[1][crow * 32 + cofs]);

  v8f acc0[4], acc1[4];
#pragma unroll
  for (int t = 0; t < 4; ++t)
#pragma unroll
    for (int e = 0; e < 8; ++e) { acc0[t][e] = 0.0f; acc1[t][e] = 0.0f; }

  const unsigned short* arow0 = x + (size_t)(row0 + rrow) * EDIM;
  const unsigned short* arow1 = arow0 + (size_t)16 * EDIM;

  async_g2l_b128(ldst0, wsrc);                       // prefetch kk=0 -> buf0

  for (int kk = 0, it = 0; kk < EDIM; kk += 32, ++it) {
    const int cur = it & 1;
    const unsigned short* bl = &bsm[cur][0];
    if (kk + 32 < EDIM) {
      async_g2l_b128(cur ? ldst0 : ldst1, wsrc + kk + 32);
      asm volatile("s_wait_asynccnt 0x1" ::: "memory");
    } else {
      asm volatile("s_wait_asynccnt 0x0" ::: "memory");
    }
    __syncthreads();                                  // slab visible to all waves

    U8 a0, a1;
    loadA(arow0 + kk, half, a0);
    loadA(arow1 + kk, half, a1);
#pragma unroll
    for (int t = 0; t < 4; ++t) {
      U8 b;
      loadB(bl + (t * 16 + rrow) * 32, half, b);
      acc0[t] = wmma_bf16(a0, b, acc0[t]);
      acc1[t] = wmma_bf16(a1, b, acc1[t]);
    }
    __syncthreads();                                  // all reads done before overwrite
  }

#pragma unroll
  for (int p = 0; p < 2; ++p) {
    v8f* acc = p ? acc1 : acc0;
#pragma unroll
    for (int t = 0; t < 4; ++t)
#pragma unroll
      for (int r = 0; r < 8; ++r) {
        int grow = row0 + p * 16 + r + 8 * half;
        int b = grow >> 11, s = grow & (SEQ - 1);
        int d = t * 16 + rrow;
        unsigned short val = f2bf(acc[t][r]);
        if (which == 2) {
          vt[((size_t)(b * NH + h) * HD + d) * SEQ + s] = val;
        } else {
          unsigned short* o = (which == 0) ? qh : kh;
          o[((size_t)(b * NH + h) * SEQ + s) * HD + d] = val;
        }
      }
  }
}

// ---------------------------------------------------------------------------
// Kernel 2: flash attention, K/V staged in LDS via async copy.
// grid=(16,32), block=256 (8 waves). wave -> one 16-query tile of one (b,h).
// ---------------------------------------------------------------------------
__global__ void __launch_bounds__(256, 1)
mha_flash_attn(const unsigned short* __restrict__ qh,
               const unsigned short* __restrict__ kh,
               const unsigned short* __restrict__ vt,
               unsigned short* __restrict__ cx) {
  __shared__ __align__(16) unsigned short pl[8][16 * 32];    // 8KB  P slabs
  __shared__ __align__(16) unsigned short ksm[2][32 * 64];   // 8KB  K slabs
  __shared__ __align__(16) unsigned short vsm[2][64 * 32];   // 8KB  V slabs

  const int tid  = threadIdx.x;
  const int lane = tid & 31, wave = tid >> 5;
  const int rrow = lane & 15, half = lane >> 4;
  const int bh = blockIdx.y;
  const int b = bh >> 4, h = bh & 15;
  const int q0 = (blockIdx.x * 8 + wave) * 16;

  const unsigned short* qhead = qh + (size_t)bh * SEQ * HD;
  const unsigned short* khead = kh + (size_t)bh * SEQ * HD;
  const unsigned short* vhead = vt + (size_t)bh * HD * SEQ;

  const int krow = tid >> 3, kofs = (tid & 7) * 8;
  const int vrow = tid >> 2, vofs = (tid & 3) * 8;
  const unsigned short* ksrc = khead + (size_t)krow * HD + kofs;
  const unsigned short* vsrc = vhead + (size_t)vrow * SEQ + vofs;
  const unsigned kd0 = lds_off(&ksm[0][krow * 64 + kofs]);
  const unsigned kd1 = lds_off(&ksm[1][krow * 64 + kofs]);
  const unsigned vd0 = lds_off(&vsm[0][vrow * 32 + vofs]);
  const unsigned vd1 = lds_off(&vsm[1][vrow * 32 + vofs]);

  U8 aq[2];
  {
    const unsigned short* qrow = qhead + (size_t)(q0 + rrow) * HD;
    loadA(qrow,      half, aq[0]);
    loadA(qrow + 32, half, aq[1]);
  }

  v8f acc[4];
#pragma unroll
  for (int t = 0; t < 4; ++t)
#pragma unroll
    for (int e = 0; e < 8; ++e) acc[t][e] = 0.0f;
  float mrow[8], lrow[8];
#pragma unroll
  for (int r = 0; r < 8; ++r) { mrow[r] = -1.0e30f; lrow[r] = 0.0f; }

  async_g2l_b128(kd0, ksrc);                    // prefetch key-block 0
  async_g2l_b128(vd0, vsrc);

  for (int kb = 0, it = 0; kb < SEQ; kb += 32, ++it) {
    const int cur = it & 1;
    const unsigned short* kl = &ksm[cur][0];
    const unsigned short* vl = &vsm[cur][0];
    if (kb + 32 < SEQ) {
      async_g2l_b128(cur ? kd0 : kd1, ksrc + (size_t)(kb + 32) * HD);
      async_g2l_b128(cur ? vd0 : vd1, vsrc + (kb + 32));
      asm volatile("s_wait_asynccnt 0x2" ::: "memory");
    } else {
      asm volatile("s_wait_asynccnt 0x0" ::: "memory");
    }
    __syncthreads();

    // ---- scores: two 16x16 tiles, K-contraction = 64
    v8f s0, s1;
#pragma unroll
    for (int e = 0; e < 8; ++e) { s0[e] = 0.0f; s1[e] = 0.0f; }
#pragma unroll
    for (int c = 0; c < 2; ++c) {
      U8 bk;
      loadB(kl + rrow * 64 + c * 32, half, bk);
      s0 = wmma_bf16(aq[c], bk, s0);
    }
#pragma unroll
    for (int c = 0; c < 2; ++c) {
      U8 bk;
      loadB(kl + (16 + rrow) * 64 + c * 32, half, bk);
      s1 = wmma_bf16(aq[c], bk, s1);
    }

    // ---- online softmax
    float pa0[8], pa1[8], alpha[8];
#pragma unroll
    for (int r = 0; r < 8; ++r) {
      float a0 = s0[r] * 0.125f;   // 1/sqrt(64)
      float a1 = s1[r] * 0.125f;
      float mx = fmaxf(a0, a1);
#pragma unroll
      for (int off = 1; off < 16; off <<= 1) mx = fmaxf(mx, __shfl_xor(mx, off, 32));
      float mnew = fmaxf(mrow[r], mx);
      alpha[r] = __expf(mrow[r] - mnew);
      mrow[r] = mnew;
      pa0[r] = __expf(a0 - mnew);
      pa1[r] = __expf(a1 - mnew);
      float sum = pa0[r] + pa1[r];
#pragma unroll
      for (int off = 1; off < 16; off <<= 1) sum += __shfl_xor(sum, off, 32);
      lrow[r] = lrow[r] * alpha[r] + sum;
    }
#pragma unroll
    for (int t = 0; t < 4; ++t)
#pragma unroll
      for (int r = 0; r < 8; ++r) acc[t][r] *= alpha[r];

    // ---- C-layout P -> LDS -> A-layout P (per-wave private slab)
    unsigned short* pw = pl[wave];
#pragma unroll
    for (int r = 0; r < 8; ++r) {
      int m = r + 8 * half;
      pw[m * 32 + rrow]      = f2bf(pa0[r]);
      pw[m * 32 + 16 + rrow] = f2bf(pa1[r]);
    }
    __asm__ volatile("s_wait_dscnt 0" ::: "memory");
    U8 ap;
    loadA(pw + rrow * 32, half, ap);

    // ---- ctx += P x V
#pragma unroll
    for (int t = 0; t < 4; ++t) {
      U8 bv;
      loadB(vl + (t * 16 + rrow) * 32, half, bv);
      acc[t] = wmma_bf16(ap, bv, acc[t]);
    }
    __syncthreads();
  }

#pragma unroll
  for (int r = 0; r < 8; ++r) {
    float inv = 1.0f / lrow[r];
    int grow = b * SEQ + q0 + r + 8 * half;
#pragma unroll
    for (int t = 0; t < 4; ++t)
      cx[(size_t)grow * EDIM + h * HD + t * 16 + rrow] = f2bf(acc[t][r] * inv);
  }
}

// ---------------------------------------------------------------------------
// Kernel 3: output projection ctx(bf16) @ Wo^T(bf16) -> fp32, Wo staged in LDS.
// grid=(16,16), block=256 (8 waves). wave -> 32 rows x 64 cols.
// ---------------------------------------------------------------------------
__global__ void __launch_bounds__(256, 1)
mha_out_proj(const unsigned short* __restrict__ cx,
             const unsigned short* __restrict__ wo,
             float* __restrict__ out) {
  __shared__ __align__(16) unsigned short bsm[2][64 * 32];

  const int tid  = threadIdx.x;
  const int lane = tid & 31, wave = tid >> 5;
  const int rrow = lane & 15, half = lane >> 4;
  const int row0 = (blockIdx.x * 8 + wave) * 32;
  const int col0 = blockIdx.y * 64;

  const int crow = tid >> 2, cofs = (tid & 3) * 8;
  const unsigned short* wsrc = wo + (size_t)(col0 + crow) * EDIM + cofs;
  const unsigned ldst0 = lds_off(&bsm[0][crow * 32 + cofs]);
  const unsigned ldst1 = lds_off(&bsm[1][crow * 32 + cofs]);

  v8f acc0[4], acc1[4];
#pragma unroll
  for (int t = 0; t < 4; ++t)
#pragma unroll
    for (int e = 0; e < 8; ++e) { acc0[t][e] = 0.0f; acc1[t][e] = 0.0f; }

  const unsigned short* arow0 = cx + (size_t)(row0 + rrow) * EDIM;
  const unsigned short* arow1 = arow0 + (size_t)16 * EDIM;

  async_g2l_b128(ldst0, wsrc);

  for (int kk = 0, it = 0; kk < EDIM; kk += 32, ++it) {
    const int cur = it & 1;
    const unsigned short* bl = &bsm[cur][0];
    if (kk + 32 < EDIM) {
      async_g2l_b128(cur ? ldst0 : ldst1, wsrc + kk + 32);
      asm volatile("s_wait_asynccnt 0x1" ::: "memory");
    } else {
      asm volatile("s_wait_asynccnt 0x0" ::: "memory");
    }
    __syncthreads();

    U8 a0, a1;
    loadA(arow0 + kk, half, a0);
    loadA(arow1 + kk, half, a1);
#pragma unroll
    for (int t = 0; t < 4; ++t) {
      U8 b;
      loadB(bl + (t * 16 + rrow) * 32, half, b);
      acc0[t] = wmma_bf16(a0, b, acc0[t]);
      acc1[t] = wmma_bf16(a1, b, acc1[t]);
    }
    __syncthreads();
  }

#pragma unroll
  for (int p = 0; p < 2; ++p) {
    v8f* acc = p ? acc1 : acc0;
#pragma unroll
    for (int t = 0; t < 4; ++t)
#pragma unroll
      for (int r = 0; r < 8; ++r)
        out[(size_t)(row0 + p * 16 + r + 8 * half) * EDIM + col0 + t * 16 + rrow] =
            acc[t][r];
  }
}

// ---------------------------------------------------------------------------
extern "C" void kernel_launch(void* const* d_in, const int* in_sizes, int n_in,
                              void* d_out, int out_size, void* d_ws, size_t ws_size,
                              hipStream_t stream) {
  (void)in_sizes; (void)n_in; (void)out_size; (void)ws_size;
  const float* q  = (const float*)d_in[0];
  const float* k  = (const float*)d_in[1];
  const float* v  = (const float*)d_in[2];
  const float* Wq = (const float*)d_in[3];
  const float* Wk = (const float*)d_in[4];
  const float* Wv = (const float*)d_in[5];
  const float* Wo = (const float*)d_in[6];
  float* out = (float*)d_out;

  const size_t E  = (size_t)NB * SEQ * EDIM;   // 4,194,304 elements
  const size_t WN = (size_t)EDIM * EDIM;       // 1,048,576 elements
  unsigned short* ws = (unsigned short*)d_ws;
  unsigned short* xq = ws;            // bf16 copies of inputs
  unsigned short* xk = xq + E;
  unsigned short* xv = xk + E;
  unsigned short* wq = xv + E;        // bf16 weights
  unsigned short* wk = wq + WN;
  unsigned short* wv = wk + WN;
  unsigned short* wo = wv + WN;
  unsigned short* qh = wo + WN;       // bf16 [B,H,S,64]
  unsigned short* kh = qh + E;        // bf16 [B,H,S,64]
  unsigned short* vt = kh + E;        // bf16 [B,H,64,S]
  unsigned short* cx = xq;            // bf16 [B,S,DIM] (aliases xq: dead by then)

  const int cvtThreads = 256;
  cvt_f32_to_bf16<<<(int)(E / (8 * cvtThreads)), cvtThreads, 0, stream>>>(q, xq, (int)E);
  cvt_f32_to_bf16<<<(int)(E / (8 * cvtThreads)), cvtThreads, 0, stream>>>(k, xk, (int)E);
  cvt_f32_to_bf16<<<(int)(E / (8 * cvtThreads)), cvtThreads, 0, stream>>>(v, xv, (int)E);
  cvt_f32_to_bf16<<<(int)(WN / (8 * cvtThreads)), cvtThreads, 0, stream>>>(Wq, wq, (int)WN);
  cvt_f32_to_bf16<<<(int)(WN / (8 * cvtThreads)), cvtThreads, 0, stream>>>(Wk, wk, (int)WN);
  cvt_f32_to_bf16<<<(int)(WN / (8 * cvtThreads)), cvtThreads, 0, stream>>>(Wv, wv, (int)WN);
  cvt_f32_to_bf16<<<(int)(WN / (8 * cvtThreads)), cvtThreads, 0, stream>>>(Wo, wo, (int)WN);

  mha_qkv_proj<<<dim3(16, 16, 3), 256, 0, stream>>>(xq, xk, xv, wq, wk, wv, qh, kh, vt);
  mha_flash_attn<<<dim3(16, 32), 256, 0, stream>>>(qh, kh, vt, cx);
  mha_out_proj<<<dim3(16, 16), 256, 0, stream>>>(cx, wo, out);
}
